// GDiffusionModel_35003983462527
// MI455X (gfx1250) — matrix operand order repaired and checked by reading, not compiled, not run on hardware
//
#include <hip/hip_runtime.h>
#include <hip/hip_bf16.h>

// ---------------------------------------------------------------------------
// Fractal diffusion model, fused persistent kernel for MI455X (gfx1250).
//   - x tile staged global->LDS with GLOBAL_LOAD_ASYNC_TO_LDS_B128 (ASYNCcnt),
//     overlapped with weight-fragment packing.
//   - z tile (64 x 256, f16) stays resident in LDS for all 99 steps.
//   - All matmuls via v_wmma_f32_16x16x32_f16 (f16 A/B, f32 accumulate).
//   - Generator weight B-fragments pre-packed in LDS in WMMA lane layout,
//     reloaded per use with ds_load_b128 (volatile -> no pressure blowup).
//   - Box-Muller pairs: one hash + log/sincos/sqrt per TWO output elements.
// ---------------------------------------------------------------------------

typedef __attribute__((ext_vector_type(16))) _Float16 v16h;
typedef __attribute__((ext_vector_type(8)))  _Float16 v8h;
typedef __attribute__((ext_vector_type(8)))  float    v8f;

#define BROWS  16384
#define DDIM   256
#define NF     32
#define MTILE  64
#define ZLD    (DDIM + 8)   // f16 units; row = 528B (16B aligned)
#define HLD    (NF + 8)     // f16 units; row = 80B  (16B aligned)
#define NFRAGS 34           // w1: 0..15 (kc*2+nh), w2: 16+nh, w3: 18+nh*8+j

__device__ __forceinline__ v8f wmma_f16(v16h a, v16h b, v8f c) {
  return __builtin_amdgcn_wmma_f32_16x16x32_f16(
      /*neg_a=*/false, a, /*neg_b=*/false, b,
      /*c_mod=*/(short)0, c, /*reuse_a=*/false, /*reuse_b=*/false);
}

// A fragment (16x32, f16) from row-major f16 LDS. ISA layout:
//   lane<16 : row=lane,    K = k0+0..7   and k0+16..23
//   lane>=16: row=lane-16, K = k0+8..15  and k0+24..31
__device__ __forceinline__ v16h load_a_lds(const _Float16* base, int m0, int k0,
                                           int ld, int lane) {
  const int r  = lane & 15;
  const int hi = (lane >> 4) * 8;
  const _Float16* p = base + (m0 + r) * ld + k0 + hi;
  union { v16h v; v8h h[2]; } u;
  u.h[0] = *(const v8h*)(p);
  u.h[1] = *(const v8h*)(p + 16);
  return u.v;
}

// A fragment from row-major f32 LDS (x staging buffer), converted to f16.
__device__ __forceinline__ v16h load_a_lds_f32(const float* base, int m0,
                                               int ld, int k0, int lane) {
  const int r  = lane & 15;
  const int hi = (lane >> 4) * 8;
  const float* p = base + (m0 + r) * ld + k0 + hi;
  v16h a;
#pragma unroll
  for (int t = 0; t < 8; ++t) {
    a[t]     = (_Float16)p[t];
    a[8 + t] = (_Float16)p[t + 16];
  }
  return a;
}

// B fragment reload from the LDS fragment cache (WMMA lane layout, 32B/lane).
// volatile: force per-use ds_load_b128 pair; never hoisted into registers.
__device__ __forceinline__ v16h load_b_lds(const _Float16* wp, int fid, int lane) {
  const volatile v8h* p = (const volatile v8h*)(wp + (fid * 32 + lane) * 16);
  union { v16h v; v8h h[2]; } u;
  u.h[0] = p[0];
  u.h[1] = p[1];
  return u.v;
}

// B fragment (32x16, f16) gathered from row-major f32 global W[K][ldn].
// Used once at setup (packing) and in the one-shot encoder.
__device__ __forceinline__ v16h load_b_gl_f32(const float* W, int ldn, int k0,
                                              int n0, int lane) {
  const int n  = n0 + (lane & 15);
  const int kh = k0 + (lane >> 4) * 8;
  v16h b;
#pragma unroll
  for (int v = 0; v < 8; ++v) {
    const int kk = kh + 2 * v + (v >= 4 ? 8 : 0);
    b[2 * v]     = (_Float16)W[(kk)     * ldn + n];
    b[2 * v + 1] = (_Float16)W[(kk + 1) * ldn + n];
  }
  return b;
}

// Counter-based hash -> two N(0,1) draws via one Box-Muller pair.
__device__ __forceinline__ float2 nrand2(unsigned idx, unsigned step) {
  unsigned x = idx * 0x9E3779B9u + step * 0x85EBCA77u + 0x165667B1u;
  x ^= x >> 16; x *= 0x7FEB352Du; x ^= x >> 15; x *= 0x846CA68Bu; x ^= x >> 16;
  unsigned y = x ^ 0x6A09E667u;
  y *= 0x9E3779B1u; y ^= y >> 15; y *= 0x85EBCA77u; y ^= y >> 13;
  float u1 = (float)(x >> 8) * 5.9604645e-08f + 5.9604645e-08f;  // (0,1]
  float u2 = (float)(y >> 8) * 5.9604645e-08f;
  float r = __builtin_sqrtf(-2.0f * __logf(u1));
  float s, c;
  __sincosf(6.2831853f * u2, &s, &c);
  return make_float2(r * c, r * s);
}

__global__ __launch_bounds__(256, 1)
void fractal_diffusion_gfx1250(const float* __restrict__ x,
                               const float* __restrict__ alpha,
                               const float* __restrict__ sigma,
                               const float* __restrict__ ew1, const float* __restrict__ eb1,
                               const float* __restrict__ ew2, const float* __restrict__ eb2,
                               const float* __restrict__ gw1, const float* __restrict__ gb1,
                               const float* __restrict__ gw2, const float* __restrict__ gb2,
                               const float* __restrict__ gw3, const float* __restrict__ gb3,
                               float* __restrict__ out) {
  __shared__ __align__(16) float    xstage[MTILE * DDIM];    // async-staged x
  __shared__ __align__(16) _Float16 zs [MTILE * ZLD];        // resident z tile
  __shared__ __align__(16) _Float16 h1s[MTILE * HLD];
  __shared__ __align__(16) _Float16 h2s[MTILE * HLD];
  __shared__ __align__(32) _Float16 wpack[NFRAGS * 32 * 16]; // B-frag cache

  const int lane = threadIdx.x & 31;
  const int w    = threadIdx.x >> 5;   // wave 0..7
  const int mi   = w & 3;              // M-tile index (m0 = mi*16)
  const int nh   = w >> 2;             // N-half for the NF=32 dimension
  const int m0   = mi * 16;
  const int ncol = lane & 15;
  const int moff = (lane >> 4) * 8;    // C/D frag: lanes 16..31 hold M+8
  const long rowbase = (long)blockIdx.x * MTILE;

  // ---- Async DMA: stage this block's x tile (64KB) into LDS ---------------
  // global_load_async_to_lds_b128: tracked with ASYNCcnt, overlaps with the
  // weight packing below; consumed by the encoder after the barrier.
  {
    const float* gsrc = x + rowbase * DDIM;
#pragma unroll
    for (int c = 0; c < 16; ++c) {
      const int elt = (threadIdx.x + c * 256) * 4;   // 4 floats / 16B chunk
      unsigned lofs = (unsigned)(size_t)&xstage[elt];
      const float* ga = gsrc + elt;
      asm volatile("global_load_async_to_lds_b128 %0, %1, off"
                   :: "v"(lofs), "v"(ga) : "memory");
    }
  }

  // ---- Pack generator weight B-fragments into LDS (once). Waves 0/4 only. --
  if (mi == 0) {
    v16h* wp = (v16h*)wpack;
#pragma unroll
    for (int kc = 0; kc < 8; ++kc)
      wp[(kc * 2 + nh) * 32 + lane] = load_b_gl_f32(gw1, NF, kc * 32, nh * 16, lane);
    wp[(16 + nh) * 32 + lane] = load_b_gl_f32(gw2, NF, 0, nh * 16, lane);
#pragma unroll
    for (int j = 0; j < 8; ++j)
      wp[(18 + nh * 8 + j) * 32 + lane] =
          load_b_gl_f32(gw3, DDIM, 0, (nh * 8 + j) * 16, lane);
  }

  // Per-lane, per-output-column state for mm3 epilogue (8 N-tiles per wave).
  int   dcol[8];
  float b3v[8], av[8], sv[8], aup[8], sdn[8];
#pragma unroll
  for (int j = 0; j < 8; ++j) {
    const int d = (nh * 8 + j) * 16 + ncol;
    dcol[j] = d;
    b3v[j]  = gb3[d];
    const float al = alpha[d], sg = sigma[d];
    av[j]  = __expf(-0.5f * al * 0.99f);   // a(i) = exp(-0.5*alpha*i/100), i=99
    sv[j]  = __expf( 0.5f * sg * 0.99f);
    aup[j] = __expf( 0.5f * al * 0.01f);   // a(i-1) = a(i)*aup
    sdn[j] = __expf(-0.5f * sg * 0.01f);
  }
  const float b1v = gb1[nh * 16 + ncol];
  const float b2v = gb2[nh * 16 + ncol];

  // Wait for the async x DMA, then make it visible to the whole workgroup.
  asm volatile("s_wait_asynccnt 0" ::: "memory");
  __syncthreads();

  // ---------------- Encoder: z = relu(x@ew1+eb1)@ew2 + eb2, diffuse(99) -----
  {
    const float e1 = eb1[nh * 16 + ncol];
    v8f acc = {};
#pragma unroll
    for (int kc = 0; kc < 8; ++kc) {
      v16h a = load_a_lds_f32(xstage, m0, DDIM, kc * 32, lane);
      v16h b = load_b_gl_f32(ew1, NF, kc * 32, nh * 16, lane);
      acc = wmma_f16(a, b, acc);
    }
#pragma unroll
    for (int v = 0; v < 8; ++v) {
      float r = fmaxf(acc[v] + e1, 0.0f);
      h1s[(m0 + v + moff) * HLD + nh * 16 + ncol] = (_Float16)r;
    }
  }
  __syncthreads();
  {
    v16h a = load_a_lds(h1s, m0, 0, HLD, lane);
#pragma unroll
    for (int j = 0; j < 8; ++j) {
      v16h b = load_b_gl_f32(ew2, DDIM, 0, (nh * 8 + j) * 16, lane);
      v8f acc = {};
      acc = wmma_f16(a, b, acc);
      const float e2 = eb2[dcol[j]];
#pragma unroll
      for (int v = 0; v < 8; v += 2) {
        const int row = m0 + v + moff;
        float z0 = acc[v]     + e2;   // encoder 2nd layer: no ReLU
        float z1 = acc[v + 1] + e2;
        const unsigned gidx = (unsigned)((rowbase + row) * DDIM + dcol[j]);
        float2 nn = nrand2(gidx, 99u);
        z0 = z0 * av[j] + sv[j] * nn.x;
        z1 = z1 * av[j] + sv[j] * nn.y;
        zs[(row)     * ZLD + dcol[j]] = (_Float16)z0;
        zs[(row + 1) * ZLD + dcol[j]] = (_Float16)z1;
      }
      av[j] *= aup[j];
      sv[j] *= sdn[j];
    }
  }
  __syncthreads();

  // ---------------- 99 generator steps, z resident in LDS -------------------
  for (int i = 99; i >= 1; --i) {
    // mm1: h1 = relu(z @ gw1 + gb1)   [64x256]x[256x32]
    v8f acc = {};
#pragma unroll
    for (int kc = 0; kc < 8; ++kc) {
      v16h a = load_a_lds(zs, m0, kc * 32, ZLD, lane);
      v16h b = load_b_lds(wpack, kc * 2 + nh, lane);
      acc = wmma_f16(a, b, acc);
    }
#pragma unroll
    for (int v = 0; v < 8; ++v) {
      float r = fmaxf(acc[v] + b1v, 0.0f);
      h1s[(m0 + v + moff) * HLD + nh * 16 + ncol] = (_Float16)r;
    }
    __syncthreads();

    // mm2: h2 = relu(h1 @ gw2 + gb2)  [64x32]x[32x32]
    {
      v16h a = load_a_lds(h1s, m0, 0, HLD, lane);
      v16h b = load_b_lds(wpack, 16 + nh, lane);
      v8f acc2 = {};
      acc2 = wmma_f16(a, b, acc2);
#pragma unroll
      for (int v = 0; v < 8; ++v) {
        float r = fmaxf(acc2[v] + b2v, 0.0f);
        h2s[(m0 + v + moff) * HLD + nh * 16 + ncol] = (_Float16)r;
      }
    }
    __syncthreads();

    // mm3: z = relu(h2 @ gw3 + gb3); fused diffuse(i-1); store
    {
      v16h a3 = load_a_lds(h2s, m0, 0, HLD, lane);
#pragma unroll
      for (int j = 0; j < 8; ++j) {
        v16h b = load_b_lds(wpack, 18 + nh * 8 + j, lane);
        v8f acc3 = {};
        acc3 = wmma_f16(a3, b, acc3);
        float zv[8];
#pragma unroll
        for (int v = 0; v < 8; v += 2) {
          const int row = m0 + v + moff;
          float t0 = fmaxf(acc3[v]     + b3v[j], 0.0f);
          float t1 = fmaxf(acc3[v + 1] + b3v[j], 0.0f);
          const unsigned gidx = (unsigned)((rowbase + row) * DDIM + dcol[j]);
          float2 nn = nrand2(gidx, (unsigned)(i - 1));
          zv[v]     = t0 * av[j] + sv[j] * nn.x;
          zv[v + 1] = t1 * av[j] + sv[j] * nn.y;
        }
        if (i > 1) {
#pragma unroll
          for (int v = 0; v < 8; ++v)
            zs[(m0 + v + moff) * ZLD + dcol[j]] = (_Float16)zv[v];
        } else {
#pragma unroll
          for (int v = 0; v < 8; ++v)
            out[(rowbase + m0 + v + moff) * DDIM + dcol[j]] = zv[v];
        }
        av[j] *= aup[j];
        sv[j] *= sdn[j];
      }
    }
    __syncthreads();
  }
}

extern "C" void kernel_launch(void* const* d_in, const int* in_sizes, int n_in,
                              void* d_out, int out_size, void* d_ws, size_t ws_size,
                              hipStream_t stream) {
  (void)in_sizes; (void)n_in; (void)out_size; (void)d_ws; (void)ws_size;
  const float* x   = (const float*)d_in[0];
  const float* al  = (const float*)d_in[1];
  const float* sg  = (const float*)d_in[2];
  const float* ew1 = (const float*)d_in[3];
  const float* eb1 = (const float*)d_in[4];
  const float* ew2 = (const float*)d_in[5];
  const float* eb2 = (const float*)d_in[6];
  const float* gw1 = (const float*)d_in[7];
  const float* gb1 = (const float*)d_in[8];
  const float* gw2 = (const float*)d_in[9];
  const float* gb2 = (const float*)d_in[10];
  const float* gw3 = (const float*)d_in[11];
  const float* gb3 = (const float*)d_in[12];

  dim3 grid(BROWS / MTILE);   // 256 workgroups
  dim3 block(256);            // 8 wave32 waves
  fractal_diffusion_gfx1250<<<grid, block, 0, stream>>>(
      x, al, sg, ew1, eb1, ew2, eb2, gw1, gb1, gw2, gb2, gw3, gb3,
      (float*)d_out);
}
